// MultiHeadAttention_48335561949316
// MI455X (gfx1250) — matrix operand order repaired and checked by reading, not compiled, not run on hardware
//
#include <hip/hip_runtime.h>
#include <hip/hip_bf16.h>
#include <stdint.h>

// ---------------------------------------------------------------------------
// MHA block for MI455X (gfx1250): bf16 WMMA everywhere, f32 accumulation.
// HBM-bound on the 1GB weights tensor -> stream it; GEMMs tiled so each wave
// runs 8 back-to-back v_wmma per s_wait (2 m-tiles x 4 n-tiles).
// ---------------------------------------------------------------------------

typedef __attribute__((ext_vector_type(16))) __bf16 v16bf;
typedef __attribute__((ext_vector_type(8)))  float  v8f;

#define S_LEN 2048
#define EMB   512
#define HEADS 8
#define DHEAD 64
#define MROWS 16384   // B*S

union FragBF { v16bf v; unsigned u[8]; };

__device__ __forceinline__ unsigned short f32_to_bf16(float f) {
  union { float f; unsigned u; } c; c.f = f;
  unsigned u = c.u;
  return (unsigned short)((u + 0x7FFFu + ((u >> 16) & 1u)) >> 16);
}
__device__ __forceinline__ unsigned pack_bf16(float a, float b) {
  return (unsigned)f32_to_bf16(a) | ((unsigned)f32_to_bf16(b) << 16);
}

__device__ __forceinline__ v8f wmma_bf16(const v16bf& a, const v16bf& b, v8f c) {
  return __builtin_amdgcn_wmma_f32_16x16x32_bf16(false, a, false, b, (short)0, c,
                                                 false, false);
}

// ===========================================================================
// Kernel 1: generic projection GEMM.  Y = A(f32, Mx512) * W(f32, 512x512) + b
//   out_mode 0: write bf16 at (B,H,S,64)      (q_s / k_s)
//   out_mode 1: write bf16 at (B,H,64,S)      (v transposed, for PV B-frags)
//   out_mode 2: write f32  at (M,512) + residual (output projection)
// 4 waves/WG; each wave = 2 m-tiles (32 rows) x 4 n-tiles (64 cols).
// K chunked 128 through LDS (48 KB of the 320 KB WGP pool).
// ===========================================================================
#define BM 128
#define BN 64
#define KC 128

__global__ __launch_bounds__(128) void proj_gemm(
    const float* __restrict__ A, const float* __restrict__ W,
    const float* __restrict__ bias, const float* __restrict__ residual,
    void* __restrict__ out, int out_mode) {
  __shared__ unsigned short sA[BM * KC];   // [row][k]      32 KB
  __shared__ unsigned short sW[BN * KC];   // [n][k] (W^T)  16 KB

  const int tid  = threadIdx.x;
  const int lane = tid & 31;
  const int wave = tid >> 5;
  const int row0 = blockIdx.x * BM;
  const int n0   = blockIdx.y * BN;

  v8f acc[2][4] = {};   // [m-tile][n-tile] 16x16 f32

  const int nloc = lane & 15;
  const int hiA  = (lane & 16) ? 8  : 0;   // A-frag k base (ISA 16b A layout)
  const int hiB  = (lane & 16) ? 16 : 0;   // unused: B-frags use hiA, same k order as A

  for (int kb = 0; kb < EMB / KC; ++kb) {
    // --- stage A chunk (f32 -> bf16), 128x128, float2-vectorized ---
    for (int i = tid; i < BM * (KC / 2); i += 128) {
      int r  = i / (KC / 2);
      int c2 = i % (KC / 2);
      const float2 f =
          *(const float2*)(A + (size_t)(row0 + r) * EMB + kb * KC + c2 * 2);
      *(unsigned*)(&sA[r * KC + c2 * 2]) = pack_bf16(f.x, f.y);
    }
    // --- stage W chunk transposed: sW[n][k] = W[kb*KC+k][n0+n] ---
    for (int i = tid; i < BN * KC; i += 128) {
      int k = i / BN;
      int n = i % BN;   // consecutive tid -> consecutive n: coalesced reads
      sW[n * KC + k] = f32_to_bf16(W[(size_t)(kb * KC + k) * EMB + n0 + n]);
    }
    if (kb < EMB / KC - 1) {   // gfx1250 global_prefetch_b8 of next chunk
      __builtin_prefetch(&A[(size_t)(row0 + tid) * EMB + (kb + 1) * KC], 0, 3);
      __builtin_prefetch(&W[(size_t)((kb + 1) * KC + tid) * EMB + n0], 0, 3);
    }
    __syncthreads();

    const int arow0 = (wave * 32 + (lane & 15)) * KC;
    const int arow1 = arow0 + 16 * KC;
#pragma unroll
    for (int ks = 0; ks < KC / 32; ++ks) {
      const int k0 = ks * 32;
      // batch ALL fragment loads, then a dense burst of 8 WMMAs
      FragBF a0, a1, bf[4];
#pragma unroll
      for (int p = 0; p < 4; ++p) {
        a0.u[p]     = *(const unsigned*)(&sA[arow0 + k0 + hiA + 2 * p]);
        a0.u[4 + p] = *(const unsigned*)(&sA[arow0 + k0 + hiA + 16 + 2 * p]);
        a1.u[p]     = *(const unsigned*)(&sA[arow1 + k0 + hiA + 2 * p]);
        a1.u[4 + p] = *(const unsigned*)(&sA[arow1 + k0 + hiA + 16 + 2 * p]);
      }
#pragma unroll
      for (int nt = 0; nt < 4; ++nt) {
        const int bcol = (nt * 16 + nloc) * KC;
#pragma unroll
        for (int p = 0; p < 8; ++p)
          bf[nt].u[p] = *(const unsigned*)(&sW[bcol + k0 + hiA + 2 * p + (p >= 4 ? 8 : 0)]);
      }
#pragma unroll
      for (int nt = 0; nt < 4; ++nt) {
        acc[0][nt] = wmma_bf16(a0.v, bf[nt].v, acc[0][nt]);
        acc[1][nt] = wmma_bf16(a1.v, bf[nt].v, acc[1][nt]);
      }
    }
    __syncthreads();
  }

  // --- epilogue: bias, layout-specific store ---
  const int mbase = (lane & 16) ? 8 : 0;
#pragma unroll
  for (int mt = 0; mt < 2; ++mt) {
#pragma unroll
    for (int nt = 0; nt < 4; ++nt) {
      const int col = n0 + nt * 16 + nloc;
      const float bv = bias[col];
#pragma unroll
      for (int i = 0; i < 8; ++i) {
        const int r = row0 + wave * 32 + mt * 16 + mbase + i;
        float val = acc[mt][nt][i] + bv;
        const int b = r >> 11, s = r & (S_LEN - 1);
        const int h = col >> 6, d = col & (DHEAD - 1);
        if (out_mode == 0) {        // (B,H,S,64) bf16
          ((unsigned short*)out)[((((size_t)b * HEADS + h) * S_LEN + s) << 6) + d] =
              f32_to_bf16(val);
        } else if (out_mode == 1) { // (B,H,64,S) bf16  (v transposed)
          ((unsigned short*)out)[(((size_t)b * HEADS + h) * DHEAD + d) * S_LEN + s] =
              f32_to_bf16(val);
        } else {                    // (M,512) f32 + residual
          val += residual[(size_t)r * EMB + col];
          ((float*)out)[(size_t)r * EMB + col] = val;
        }
      }
    }
  }
}

// ===========================================================================
// Kernel 2: scores = q_s @ k_s^T / 8, masked; f32 straight into d_out weights.
// Per WG: one (b,h), 64x64 tile. K=64 -> both k-steps' fragments preloaded,
// then 8 back-to-back WMMAs. q_s/k_s (32MB) live in the 192MB L2.
// ===========================================================================
__global__ __launch_bounds__(128) void attn_scores(
    const unsigned short* __restrict__ qs, const unsigned short* __restrict__ ks,
    const unsigned char* __restrict__ mask, float* __restrict__ wts) {
  const int tid  = threadIdx.x;
  const int lane = tid & 31;
  const int wave = tid >> 5;
  const int bh   = blockIdx.z;
  const int q0   = blockIdx.x * 64 + wave * 16;
  const int key0 = blockIdx.y * 64;

  const unsigned short* qh = qs + (size_t)bh * S_LEN * DHEAD;
  const unsigned short* kh = ks + (size_t)bh * S_LEN * DHEAD;

  const int nloc = lane & 15;
  const int hiA  = (lane & 16) ? 8  : 0;
  const int hiB  = (lane & 16) ? 16 : 0;

  const unsigned short* arow = qh + (size_t)(q0 + (lane & 15)) * DHEAD;

  FragBF af[2], bf[2][4];
#pragma unroll
  for (int ksx = 0; ksx < 2; ++ksx) {
    const int k0 = ksx * 32;
#pragma unroll
    for (int p = 0; p < 4; ++p) {
      af[ksx].u[p]     = *(const unsigned*)(arow + k0 + hiA + 2 * p);
      af[ksx].u[4 + p] = *(const unsigned*)(arow + k0 + hiA + 16 + 2 * p);
    }
#pragma unroll
    for (int nt = 0; nt < 4; ++nt) {
      const unsigned short* brow = kh + (size_t)(key0 + nt * 16 + nloc) * DHEAD;
#pragma unroll
      for (int p = 0; p < 8; ++p)
        bf[ksx][nt].u[p] = *(const unsigned*)(brow + k0 + hiA + 2 * p + (p >= 4 ? 8 : 0));
    }
  }

  v8f acc[4] = {};
#pragma unroll
  for (int ksx = 0; ksx < 2; ++ksx)
#pragma unroll
    for (int nt = 0; nt < 4; ++nt)
      acc[nt] = wmma_bf16(af[ksx].v, bf[ksx][nt].v, acc[nt]);

  const int b     = bh >> 3;
  const int mbase = (lane & 16) ? 8 : 0;
#pragma unroll
  for (int nt = 0; nt < 4; ++nt) {
    const int key = key0 + nt * 16 + nloc;
#pragma unroll
    for (int i = 0; i < 8; ++i) {
      const int q = q0 + mbase + i;
      float v = acc[nt][i] * 0.125f;   // 1/sqrt(64)
      if (mask[((size_t)b * S_LEN + q) * S_LEN + key]) v = -1.0e9f;
      wts[((size_t)bh * S_LEN + q) * S_LEN + key] = v;
    }
  }
}

// ===========================================================================
// Kernel 3: row softmax, in place on the weights region of d_out.
// ===========================================================================
__global__ __launch_bounds__(256) void softmax_rows(float* __restrict__ wts) {
  __shared__ float red[256];
  const int tid = threadIdx.x;
  float* p = wts + (size_t)blockIdx.x * S_LEN;

  float v[8];
#pragma unroll
  for (int j = 0; j < 8; ++j) v[j] = p[tid + j * 256];

  float m = v[0];
#pragma unroll
  for (int j = 1; j < 8; ++j) m = fmaxf(m, v[j]);
  red[tid] = m; __syncthreads();
  for (int s = 128; s > 0; s >>= 1) {
    if (tid < s) red[tid] = fmaxf(red[tid], red[tid + s]);
    __syncthreads();
  }
  const float rowmax = red[0]; __syncthreads();

  float sum = 0.f;
#pragma unroll
  for (int j = 0; j < 8; ++j) { v[j] = __expf(v[j] - rowmax); sum += v[j]; }
  red[tid] = sum; __syncthreads();
  for (int s = 128; s > 0; s >>= 1) {
    if (tid < s) red[tid] += red[tid + s];
    __syncthreads();
  }
  const float inv = 1.0f / red[0];
#pragma unroll
  for (int j = 0; j < 8; ++j) p[tid + j * 256] = v[j] * inv;
}

// ===========================================================================
// Kernel 4: context = weights @ v_s, written as (B,S,H*64) f32 for the output
// projection. This kernel streams the 1GB weights tensor (the HBM roofline),
// so the A-operand gets global_prefetch. A-frags: f32 -> bf16 on the fly;
// B-frags: packed b32 from v^T (k-pairs contiguous by design). 2 m-tiles/wave.
// ===========================================================================
__global__ __launch_bounds__(128) void attn_context(
    const float* __restrict__ wts, const unsigned short* __restrict__ vT,
    float* __restrict__ ctx) {
  const int tid  = threadIdx.x;
  const int lane = tid & 31;
  const int wave = tid >> 5;
  const int bh   = blockIdx.z;
  const int q0   = blockIdx.x * 128 + wave * 32;

  const int nloc = lane & 15;
  const int hiA  = (lane & 16) ? 8  : 0;
  const int hiB  = (lane & 16) ? 16 : 0;

  const float* arow0 = wts + ((size_t)bh * S_LEN + (q0 + (lane & 15))) * S_LEN;
  const float* arow1 = arow0 + (size_t)16 * S_LEN;
  const unsigned short* vbase = vT + (size_t)bh * DHEAD * S_LEN;

  v8f acc[2][4] = {};
  for (int kk = 0; kk < S_LEN / 32; ++kk) {
    const int k0 = kk * 32;
    if (kk + 1 < S_LEN / 32) {   // prefetch the streaming operand
      __builtin_prefetch(arow0 + k0 + 32, 0, 1);
      __builtin_prefetch(arow1 + k0 + 32, 0, 1);
    }
    FragBF a0, a1, bf[4];
#pragma unroll
    for (int p = 0; p < 4; ++p) {
      a0.u[p]     = pack_bf16(arow0[k0 + hiA + 2 * p],      arow0[k0 + hiA + 2 * p + 1]);
      a0.u[4 + p] = pack_bf16(arow0[k0 + hiA + 16 + 2 * p], arow0[k0 + hiA + 17 + 2 * p]);
      a1.u[p]     = pack_bf16(arow1[k0 + hiA + 2 * p],      arow1[k0 + hiA + 2 * p + 1]);
      a1.u[4 + p] = pack_bf16(arow1[k0 + hiA + 16 + 2 * p], arow1[k0 + hiA + 17 + 2 * p]);
    }
#pragma unroll
    for (int nt = 0; nt < 4; ++nt) {
      const unsigned short* brow = vbase + (size_t)(nt * 16 + nloc) * S_LEN;
#pragma unroll
      for (int p = 0; p < 8; ++p)
        bf[nt].u[p] = *(const unsigned*)(brow + k0 + hiA + 2 * p + (p >= 4 ? 8 : 0));
    }
#pragma unroll
    for (int nt = 0; nt < 4; ++nt) {
      acc[0][nt] = wmma_bf16(a0.v, bf[nt].v, acc[0][nt]);
      acc[1][nt] = wmma_bf16(a1.v, bf[nt].v, acc[1][nt]);
    }
  }

  const int b = bh >> 3, h = bh & 7;
  const int mbase = (lane & 16) ? 8 : 0;
#pragma unroll
  for (int mt = 0; mt < 2; ++mt) {
#pragma unroll
    for (int nt = 0; nt < 4; ++nt) {
      const int d = nt * 16 + nloc;
#pragma unroll
      for (int i = 0; i < 8; ++i) {
        const int q = q0 + mt * 16 + mbase + i;
        ctx[((size_t)b * S_LEN + q) * EMB + h * DHEAD + d] = acc[mt][nt][i];
      }
    }
  }
}

// ===========================================================================
// Kernel 5: LayerNorm of (residual + proj) rows -> d_out output region.
// ===========================================================================
__global__ __launch_bounds__(256) void resid_ln(
    const float* __restrict__ y, const float* __restrict__ gamma,
    const float* __restrict__ beta, float* __restrict__ outp) {
  __shared__ float red[256];
  const int tid = threadIdx.x;
  const float* p = y + (size_t)blockIdx.x * EMB;
  const float x0 = p[tid], x1 = p[tid + 256];

  red[tid] = x0 + x1; __syncthreads();
  for (int s = 128; s > 0; s >>= 1) {
    if (tid < s) red[tid] += red[tid + s];
    __syncthreads();
  }
  const float mu = red[0] * (1.0f / EMB); __syncthreads();

  red[tid] = x0 * x0 + x1 * x1; __syncthreads();
  for (int s = 128; s > 0; s >>= 1) {
    if (tid < s) red[tid] += red[tid + s];
    __syncthreads();
  }
  const float var = red[0] * (1.0f / EMB) - mu * mu;
  const float rs  = __frsqrt_rn(var + 1e-5f);

  float* o = outp + (size_t)blockIdx.x * EMB;
  o[tid]       = (x0 - mu) * rs * gamma[tid]       + beta[tid];
  o[tid + 256] = (x1 - mu) * rs * gamma[tid + 256] + beta[tid + 256];
}

// ===========================================================================
extern "C" void kernel_launch(void* const* d_in, const int* in_sizes, int n_in,
                              void* d_out, int out_size, void* d_ws, size_t ws_size,
                              hipStream_t stream) {
  const float*         Q     = (const float*)d_in[0];
  const float*         K     = (const float*)d_in[1];
  const float*         V     = (const float*)d_in[2];
  const unsigned char* mask  = (const unsigned char*)d_in[3];
  const float*         Wq    = (const float*)d_in[4];
  const float*         bq    = (const float*)d_in[5];
  const float*         Wo    = (const float*)d_in[6];
  const float*         bo    = (const float*)d_in[7];
  const float*         gamma = (const float*)d_in[8];
  const float*         beta  = (const float*)d_in[9];

  // d_out = [output (B,S,512) f32 | weights (B,H,S,S) f32]
  float* out_ln = (float*)d_out;
  float* wts    = out_ln + (size_t)MROWS * EMB;

  // workspace layout
  char* ws = (char*)d_ws;
  unsigned short* qs = (unsigned short*)ws;                    ws += (size_t)MROWS * EMB * 2;
  unsigned short* kk = (unsigned short*)ws;                    ws += (size_t)MROWS * EMB * 2;
  unsigned short* vT = (unsigned short*)ws;                    ws += (size_t)MROWS * EMB * 2;
  float*          ctx = (float*)ws;                            ws += (size_t)MROWS * EMB * 4;
  float*          yb  = (float*)ws;

  const dim3 blk(128);
  const dim3 gproj(MROWS / BM, EMB / BN);            // 128 x 8

  proj_gemm<<<gproj, blk, 0, stream>>>(Q, Wq, bq, nullptr, qs, 0);
  proj_gemm<<<gproj, blk, 0, stream>>>(K, Wq, bq, nullptr, kk, 0);
  proj_gemm<<<gproj, blk, 0, stream>>>(V, Wq, bq, nullptr, vT, 1);

  attn_scores<<<dim3(S_LEN / 64, S_LEN / 64, 64), blk, 0, stream>>>(qs, kk, mask, wts);
  softmax_rows<<<dim3(64 * S_LEN), 256, 0, stream>>>(wts);
  attn_context<<<dim3(S_LEN / 128, 1, 64), blk, 0, stream>>>(wts, vT, ctx);

  proj_gemm<<<gproj, blk, 0, stream>>>(ctx, Wo, bo, Q, yb, 2);
  resid_ln<<<MROWS, 256, 0, stream>>>(yb, gamma, beta, out_ln);
}
